// PositionAttention_41016937677189
// MI455X (gfx1250) — compile-verified, matrix operand order checked by reading, not run on hardware
//
#include <hip/hip_runtime.h>

// PositionAttention fused for MI455X (gfx1250, wave32, WMMA).
//
// Cost model: out = v @ softmax(q@k) is 34.4 GFLOP (f16 WMMA, ~7-13us);
// minimal HBM traffic ~85 MB (~3.7us @ 23.3 TB/s); softmax exp is ~134M
// v_exp_f32 (~13us) -> every P element must be exp'd exactly ONCE per pass.
// Never materialize the 268 MB energy matrix: pass 1 computes per-row (max,
// 1/Z) with WMMA (max sweep + one-exp sum sweep), pass 2 recomputes 16x16
// energy tiles (Cq=32 == one WMMA K step), exps in-lane, and repacks the f32
// D-fragment directly into the f16 B-fragment of the output GEMM (the layouts
// are mirror images -> zero shuffles, zero LDS, zero barriers).
// All projections are WMMA GEMMs too (x1/x2 read exactly once each).

typedef __attribute__((ext_vector_type(16))) _Float16 v16h;
typedef __attribute__((ext_vector_type(8)))  _Float16 v8h;
typedef __attribute__((ext_vector_type(8)))  float    v8f;
typedef __attribute__((ext_vector_type(4)))  float    v4f;

#define B_   8
#define C_   256
#define CQ_  32
#define P1_  2048
#define P2_  4096

__device__ __forceinline__ v8f wmma_f16(v16h a, v16h b, v8f c) {
  return __builtin_amdgcn_wmma_f32_16x16x32_f16(false, a, false, b,
                                                (short)0, c, false, false);
}

// Load a 16x32 f16 WMMA A/B fragment from a row-major [rows x ld] f16 array.
// 16-bit 16x32 layout: lane<16 -> K 0..7 & 16..23 of row (lane&15);
// lane>=16 -> K 8..15 & 24..31. Two 16B loads per lane.
__device__ __forceinline__ v16h load_frag16(const _Float16* __restrict__ base,
                                            int row, int ld, int k0, int lane) {
  const int sel = lane >> 4;
  const int r   = lane & 15;
  const _Float16* p = base + (size_t)(row + r) * (size_t)ld + k0 + sel * 8;
  v16h f;
  *(v8h*)&f       = *(const v8h*)(p);
  *((v8h*)&f + 1) = *(const v8h*)(p + 16);
  return f;
}

// ---- tiny f32 -> f16 weight conversion (one-time, KB-sized) ----
__global__ void cvt_f16(const float* __restrict__ src, _Float16* __restrict__ dst, int n) {
  const int i = blockIdx.x * blockDim.x + threadIdx.x;
  if (i < n) dst[i] = (_Float16)src[i];
}

// ---- WMMA projection GEMM: OUT = Wh[Co x 256] @ X[b][256 x P] + bias ----
// One wave per (b, 16-wide p-tile); builds the X-side B fragment once per
// K-chunk (X read exactly once), loops weight A fragments over CT c-tiles.
// OUT_CP: true -> out[b][c][p] (v layout); false -> out[b][p][Co] (q,k layout)
template <int CT, bool OUT_CP>
__global__ void proj_wmma(const _Float16* __restrict__ Wh,
                          const float* __restrict__ bias,
                          const float* __restrict__ X,
                          _Float16* __restrict__ out, int P) {
  const int lane = threadIdx.x & 31;
  const int wid  = threadIdx.x >> 5;
  const int sel  = lane >> 4;
  const int col  = lane & 15;
  constexpr int Co = CT * 16;

  const int gw = blockIdx.x * (256 >> 5) + wid;   // global wave id
  const int ptiles = P >> 4;
  const int pt = gw % ptiles;
  const int b  = gw / ptiles;
  const int pbase = pt << 4;
  const float* xp = X + (size_t)b * C_ * P + pbase + col;

  v8f acc[CT];
  #pragma unroll
  for (int t = 0; t < CT; ++t) acc[t] = (v8f){};

  for (int k0 = 0; k0 < C_; k0 += 32) {
    // B fragment: elem j = X[k0 + 8*sel + j][p]; elem 8+j = X[k0+16+8*sel+j][p]
    v16h bf;
    #pragma unroll
    for (int j = 0; j < 8; ++j) {
      bf[j]     = (_Float16)xp[(size_t)(k0 + sel * 8 + j) * P];
      bf[8 + j] = (_Float16)xp[(size_t)(k0 + 16 + sel * 8 + j) * P];
    }
    #pragma unroll
    for (int t = 0; t < CT; ++t) {
      const v16h af = load_frag16(Wh, t * 16, C_, k0, lane);
      acc[t] = wmma_f16(af, bf, acc[t]);
    }
  }
  #pragma unroll
  for (int t = 0; t < CT; ++t) {
    const int cb = t * 16 + sel * 8;               // D row = r + 8*sel
    const v4f b0 = *(const v4f*)(bias + cb);
    const v4f b1 = *(const v4f*)(bias + cb + 4);
    #pragma unroll
    for (int r = 0; r < 8; ++r) {
      const int c = cb + r;
      const float val = acc[t][r] + (r < 4 ? b0[r] : b1[r - 4]);
      if (OUT_CP)
        out[((size_t)b * Co + c) * P + pbase + col] = (_Float16)val;
      else
        out[((size_t)b * P + pbase + col) * Co + c] = (_Float16)val;
    }
  }
}

// ---- pass 1: per-p1-row softmax stats over p2 ----
// Two sweeps: max-only (no exp), then sum with ONE exp per element.
__global__ void attn_stats(const _Float16* __restrict__ qw,
                           const _Float16* __restrict__ kw,
                           float* __restrict__ mArr, float* __restrict__ zArr) {
  const int lane = threadIdx.x & 31;
  const int wid  = threadIdx.x >> 5;
  const int b    = blockIdx.y;
  const int p1base = (blockIdx.x * 8 + wid) * 16;
  const _Float16* qb = qw + (size_t)b * P1_ * CQ_;
  const _Float16* kb = kw + (size_t)b * P2_ * CQ_;

  const v16h a = load_frag16(qb, p1base, CQ_, 0, lane);

  float m[8];
  #pragma unroll
  for (int r = 0; r < 8; ++r) m[r] = -3.0e38f;

  for (int p2 = 0; p2 < P2_; p2 += 16) {
    const v16h bf = load_frag16(kb, p2, CQ_, 0, lane);
    v8f e = {};
    e = wmma_f16(a, bf, e);
    #pragma unroll
    for (int r = 0; r < 8; ++r) m[r] = fmaxf(m[r], e[r]);
  }
  #pragma unroll
  for (int off = 1; off < 16; off <<= 1)
    #pragma unroll
    for (int r = 0; r < 8; ++r) m[r] = fmaxf(m[r], __shfl_xor(m[r], off));

  float s[8];
  #pragma unroll
  for (int r = 0; r < 8; ++r) s[r] = 0.0f;

  for (int p2 = 0; p2 < P2_; p2 += 16) {
    const v16h bf = load_frag16(kb, p2, CQ_, 0, lane);
    v8f e = {};
    e = wmma_f16(a, bf, e);
    #pragma unroll
    for (int r = 0; r < 8; ++r) s[r] += __expf(e[r] - m[r]);
  }
  #pragma unroll
  for (int off = 1; off < 16; off <<= 1)
    #pragma unroll
    for (int r = 0; r < 8; ++r) s[r] += __shfl_xor(s[r], off);

  if ((lane & 15) == 0) {
    const int rowbase = p1base + (lane >> 4) * 8;   // D layout: row = r + 8*sel
    #pragma unroll
    for (int r = 0; r < 8; ++r) {
      mArr[(size_t)b * P1_ + rowbase + r] = m[r];
      zArr[(size_t)b * P1_ + rowbase + r] = 1.0f / s[r];
    }
  }
}

// ---- pass 2: out[b,c,p2] = alpha * (v @ P) + x2, P recomputed on the fly ----
// One block = 8 waves = [256 C x 128 p2] tile: wave w owns p2 strip w (16
// cols) and ALL 16 C sub-tiles, so every P element is exp'd exactly once.
// Per 32-wide p1 chunk: 2 energy WMMAs -> exp/scale in-lane -> f16 B fragment
// -> 16 accumulate WMMAs. No LDS, no barriers.
__global__ void attn_out(const _Float16* __restrict__ qw,
                         const _Float16* __restrict__ kw,
                         const _Float16* __restrict__ vw,
                         const float* __restrict__ mArr,
                         const float* __restrict__ zArr,
                         const float* __restrict__ x2,
                         const float* __restrict__ alphaPtr,
                         float* __restrict__ out) {
  const int lane  = threadIdx.x & 31;
  const int wid   = threadIdx.x >> 5;
  const int sel   = lane >> 4;
  const int col   = lane & 15;
  const int b     = blockIdx.y;
  const int strip = blockIdx.x * 128 + wid * 16;

  const _Float16* qb = qw + (size_t)b * P1_ * CQ_;
  const _Float16* kb = kw + (size_t)b * P2_ * CQ_;
  const _Float16* vb = vw + (size_t)b * C_ * P1_;
  const float* mb = mArr + (size_t)b * P1_;
  const float* zb = zArr + (size_t)b * P1_;

  const v16h kf = load_frag16(kb, strip, CQ_, 0, lane);  // chunk-invariant

  v8f acc[16];
  #pragma unroll
  for (int t = 0; t < 16; ++t) acc[t] = (v8f){};

  for (int k0 = 0; k0 < P1_; k0 += 32) {
    // energy for p1 rows [k0, k0+32) x this strip's 16 p2 cols
    const v16h alo = load_frag16(qb, k0,      CQ_, 0, lane);
    const v16h ahi = load_frag16(qb, k0 + 16, CQ_, 0, lane);
    v8f elo = {}, ehi = {};
    elo = wmma_f16(alo, kf, elo);
    ehi = wmma_f16(ahi, kf, ehi);

    // lane holds rows k0 + 8*sel + r (lo) and +16 (hi)
    const int rlo = k0 + sel * 8;
    const v4f m0 = *(const v4f*)(mb + rlo);
    const v4f m1 = *(const v4f*)(mb + rlo + 4);
    const v4f z0 = *(const v4f*)(zb + rlo);
    const v4f z1 = *(const v4f*)(zb + rlo + 4);
    const v4f m2 = *(const v4f*)(mb + rlo + 16);
    const v4f m3 = *(const v4f*)(mb + rlo + 20);
    const v4f z2 = *(const v4f*)(zb + rlo + 16);
    const v4f z3 = *(const v4f*)(zb + rlo + 20);

    // D-fragment (row=r+8*sel, col=lane%16) repacks in-lane into the f16
    // B-fragment (col=lane%16, K packed into regs) of the output GEMM.
    v16h pf;
    #pragma unroll
    for (int r = 0; r < 4; ++r) {
      pf[r]      = (_Float16)(__expf(elo[r]     - m0[r]) * z0[r]);
      pf[4 + r]  = (_Float16)(__expf(elo[4 + r] - m1[r]) * z1[r]);
      pf[8 + r]  = (_Float16)(__expf(ehi[r]     - m2[r]) * z2[r]);
      pf[12 + r] = (_Float16)(__expf(ehi[4 + r] - m3[r]) * z3[r]);
    }

    #pragma unroll
    for (int t = 0; t < 16; ++t) {
      const v16h af = load_frag16(vb, 16 * t, P1_, k0, lane);
      acc[t] = wmma_f16(af, pf, acc[t]);
    }
  }

  const float alpha = alphaPtr[0];
  #pragma unroll
  for (int t = 0; t < 16; ++t) {
    #pragma unroll
    for (int r = 0; r < 8; ++r) {
      const int c  = 16 * t + r + 8 * sel;
      const int p2 = strip + col;
      const size_t idx = ((size_t)b * C_ + c) * P2_ + p2;
      out[idx] = alpha * acc[t][r] + x2[idx];
    }
  }
}

extern "C" void kernel_launch(void* const* d_in, const int* in_sizes, int n_in,
                              void* d_out, int out_size, void* d_ws, size_t ws_size,
                              hipStream_t stream) {
  const float* x1    = (const float*)d_in[0];
  const float* x2    = (const float*)d_in[1];
  const float* Wq    = (const float*)d_in[2];
  const float* bq    = (const float*)d_in[3];
  const float* Wk    = (const float*)d_in[4];
  const float* bk    = (const float*)d_in[5];
  const float* Wv    = (const float*)d_in[6];
  const float* bv    = (const float*)d_in[7];
  const float* alpha = (const float*)d_in[8];
  float* out = (float*)d_out;

  // workspace (~11.4 MB; L2-resident on the 192 MB L2):
  char* ws = (char*)d_ws;
  _Float16* qw  = (_Float16*)(ws);                        // [8][2048][32] : 1 MB
  _Float16* kw  = (_Float16*)(ws + (1u << 20));           // [8][4096][32] : 2 MB
  _Float16* vw  = (_Float16*)(ws + (3u << 20));           // [8][256][2048]: 8 MB
  float*  mArr  = (float*)(ws + (11u << 20));             // [8][2048] : 64 KB
  float*  zArr  = (float*)(ws + (11u << 20) + (1u << 16));// [8][2048] : 64 KB
  _Float16* Wqh = (_Float16*)(ws + (11u << 20) + (2u << 16)); // 16 KB
  _Float16* Wkh = (_Float16*)(ws + (11u << 20) + (2u << 16) + (16u << 10));
  _Float16* Wvh = (_Float16*)(ws + (11u << 20) + (2u << 16) + (32u << 10)); // 128 KB

  // one-time weight conversion to f16
  cvt_f16<<<(CQ_ * C_ + 255) / 256, 256, 0, stream>>>(Wq, Wqh, CQ_ * C_);
  cvt_f16<<<(CQ_ * C_ + 255) / 256, 256, 0, stream>>>(Wk, Wkh, CQ_ * C_);
  cvt_f16<<<(C_ * C_ + 255) / 256, 256, 0, stream>>>(Wv, Wvh, C_ * C_);

  // WMMA projections: X read exactly once each
  proj_wmma<2,  false><<<(B_ * (P1_ / 16)) / 8, 256, 0, stream>>>(Wqh, bq, x1, qw, P1_);
  proj_wmma<2,  false><<<(B_ * (P2_ / 16)) / 8, 256, 0, stream>>>(Wkh, bk, x2, kw, P2_);
  proj_wmma<16, true ><<<(B_ * (P1_ / 16)) / 8, 256, 0, stream>>>(Wvh, bv, x1, vw, P1_);

  // pass 1: softmax row stats (8 waves/block, one 16-row p1 tile per wave)
  attn_stats<<<dim3(P1_ / (16 * 8), B_), 256, 0, stream>>>(qw, kw, mArr, zArr);

  // pass 2: fused energy-recompute + softmax-apply + v@P + epilogue
  attn_out<<<dim3(P2_ / 128, B_), 256, 0, stream>>>(
      qw, kw, vw, mArr, zArr, x2, alpha, out);
}
